// TopKRouter_13486197310136
// MI455X (gfx1250) — matrix operand order repaired and checked by reading, not compiled, not run on hardware
//
#include <hip/hip_runtime.h>
#include <hip/hip_bf16.h>
#include <math.h>

#define NUM_EXPERTS 64
#define HDIM 4096
#define TOKENS_PER_BLOCK 128
#define LSTRIDE 68            // 64 + pad to avoid LDS bank conflicts
#define WB_OFFSET_BYTES 1024  // bf16 W_gate starts here inside d_ws

typedef __bf16        bf16x16 __attribute__((ext_vector_type(16)));
typedef float         v8f     __attribute__((ext_vector_type(8)));
typedef float         v4f     __attribute__((ext_vector_type(4)));
typedef unsigned int  u32x8   __attribute__((ext_vector_type(8)));

// One-time proper RNE f32->bf16 pair pack (W pre-convert kernel only).
__device__ __forceinline__ unsigned pk_bf16_rne(float a, float b) {
    unsigned ua = __builtin_bit_cast(unsigned, a);
    unsigned ub = __builtin_bit_cast(unsigned, b);
    ua += 0x7FFFu + ((ua >> 16) & 1u);
    ub += 0x7FFFu + ((ub >> 16) & 1u);
    return (ua >> 16) | (ub & 0xFFFF0000u);
}

// In-loop pack: round-half-up then one v_perm_b32 grabs both high halves.
__device__ __forceinline__ unsigned pk_bf16_fast(float a, float b) {
    unsigned ua = __builtin_bit_cast(unsigned, a) + 0x8000u;
    unsigned ub = __builtin_bit_cast(unsigned, b) + 0x8000u;
    return __builtin_amdgcn_perm(ub, ua, 0x07060302u);
}

// Raw (unpacked) per-K-chunk data held in the software pipeline.
struct RawFrag {
    v4f   a0, a1, a2, a3;   // 16 f32 of A for this lane
    u32x8 b0, b1, b2, b3;   // 4 B tiles, already bf16-packed
};

__device__ __forceinline__ RawFrag load_raw(
    const float* __restrict__ aPtr, int halfOff8,
    const unsigned short* __restrict__ w0, const unsigned short* __restrict__ w1,
    const unsigned short* __restrict__ w2, const unsigned short* __restrict__ w3)
{
    RawFrag r;
    const v4f* p0 = (const v4f*)(aPtr + halfOff8);        // K = halfOff8..+7 (halfOff8 = 16*lh)
    r.a0 = p0[0]; r.a1 = p0[1];
    const v4f* p1 = (const v4f*)(aPtr + 8 + halfOff8);    // K = 8+halfOff8..+7
    r.a2 = p1[0]; r.a3 = p1[1];
    r.b0 = *(const u32x8*)w0;
    r.b1 = *(const u32x8*)w1;
    r.b2 = *(const u32x8*)w2;
    r.b3 = *(const u32x8*)w3;
    return r;
}

__device__ __forceinline__ bf16x16 pack_a(const RawFrag& r) {
    u32x8 pk;
    pk[0] = pk_bf16_fast(r.a0[0], r.a0[1]); pk[1] = pk_bf16_fast(r.a0[2], r.a0[3]);
    pk[2] = pk_bf16_fast(r.a1[0], r.a1[1]); pk[3] = pk_bf16_fast(r.a1[2], r.a1[3]);
    pk[4] = pk_bf16_fast(r.a2[0], r.a2[1]); pk[5] = pk_bf16_fast(r.a2[2], r.a2[3]);
    pk[6] = pk_bf16_fast(r.a3[0], r.a3[1]); pk[7] = pk_bf16_fast(r.a3[2], r.a3[3]);
    return __builtin_bit_cast(bf16x16, pk);
}

#define DO_WMMA(frag)                                                                        \
    do {                                                                                     \
        bf16x16 af_ = pack_a(frag);                                                          \
        bf16x16 b0_ = __builtin_bit_cast(bf16x16, (frag).b0);                                \
        bf16x16 b1_ = __builtin_bit_cast(bf16x16, (frag).b1);                                \
        bf16x16 b2_ = __builtin_bit_cast(bf16x16, (frag).b2);                                \
        bf16x16 b3_ = __builtin_bit_cast(bf16x16, (frag).b3);                                \
        acc0 = __builtin_amdgcn_wmma_f32_16x16x32_bf16(false, af_, false, b0_, (short)0, acc0, false, false); \
        acc1 = __builtin_amdgcn_wmma_f32_16x16x32_bf16(false, af_, false, b1_, (short)0, acc1, false, false); \
        acc2 = __builtin_amdgcn_wmma_f32_16x16x32_bf16(false, af_, false, b2_, (short)0, acc2, false, false); \
        acc3 = __builtin_amdgcn_wmma_f32_16x16x32_bf16(false, af_, false, b3_, (short)0, acc3, false, false); \
    } while (0)

#define LOAD_CHUNK(K0) load_raw(aRow + (K0), hOff8, wRow0 + (K0), wRow1 + (K0), \
                                wRow2 + (K0), wRow3 + (K0))

// ---- one-time W_gate f32 -> bf16 conversion (64*4096 = 256K elements) ----
__global__ void convert_w_kernel(const float* __restrict__ W,
                                 unsigned int* __restrict__ Wb /* packed pairs */) {
    int i = blockIdx.x * blockDim.x + threadIdx.x;   // pair index
    if (i < NUM_EXPERTS * HDIM / 2) {
        Wb[i] = pk_bf16_rne(W[2 * i + 0], W[2 * i + 1]);
    }
}

__global__ void zero_ws_kernel(float* __restrict__ ws) {
    int i = threadIdx.x;
    if (i < NUM_EXPERTS * 2 + 1) ws[i] = 0.0f;
}

__global__ __launch_bounds__(256) void router_gemm_topk_kernel(
    const float* __restrict__ A,            // [N, 4096] f32
    const unsigned short* __restrict__ Wb,  // [64, 4096] bf16 (pre-converted)
    float* __restrict__ out,                // weights [N,2] | indices-as-float [N,2] | aux
    float* __restrict__ ws,                 // [64] probsum, [64] counts, [1] zsum
    int nTokens)
{
    __shared__ float lds_logits[TOKENS_PER_BLOCK * LSTRIDE];
    __shared__ float lds_mx[TOKENS_PER_BLOCK];
    __shared__ float lds_rs[TOKENS_PER_BLOCK];
    __shared__ float lds_cnt[NUM_EXPERTS];
    __shared__ float lds_z;

    const int tid  = threadIdx.x;
    const int wave = tid >> 5;
    const int lane = tid & 31;
    const int lh   = lane >> 4;
    const int lm   = lane & 15;

    if (tid < NUM_EXPERTS) lds_cnt[tid] = 0.0f;
    if (tid == 0)          lds_z = 0.0f;

    // ---------------- GEMM: logits = A x W^T via bf16 WMMA ----------------
    const int rowBase = blockIdx.x * TOKENS_PER_BLOCK + wave * 16;
    const float* __restrict__ aRow = A + (size_t)(rowBase + lm) * HDIM;
    const int hOff8 = lh * 16;
    const unsigned short* __restrict__ wRow0 = Wb + (size_t)(0 * 16 + lm) * HDIM + lh * 16;
    const unsigned short* __restrict__ wRow1 = Wb + (size_t)(1 * 16 + lm) * HDIM + lh * 16;
    const unsigned short* __restrict__ wRow2 = Wb + (size_t)(2 * 16 + lm) * HDIM + lh * 16;
    const unsigned short* __restrict__ wRow3 = Wb + (size_t)(3 * 16 + lm) * HDIM + lh * 16;

    v8f acc0 = {}, acc1 = {}, acc2 = {}, acc3 = {};

    // 2-stage software pipeline, unrolled x2 with ping-pong buffers so the
    // stage rotation is a register-name swap, not a v_mov copy.
    RawFrag fa = LOAD_CHUNK(0);
#pragma unroll 1
    for (int k0 = 32; k0 < HDIM - 32; k0 += 64) {
        RawFrag fb = LOAD_CHUNK(k0);
        if (k0 + 128 < HDIM)  // scalar-uniform guard; stream A ahead of use
            __builtin_prefetch(aRow + k0 + 96, 0, 1);
        DO_WMMA(fa);                 // consume chunk k0-32; fa's regs die
        fa = LOAD_CHUNK(k0 + 32);    // fresh load written directly into fa
        DO_WMMA(fb);                 // consume chunk k0
    }
    {   // tail: chunks HDIM-64 (in fa) and HDIM-32
        RawFrag fb = LOAD_CHUNK(HDIM - 32);
        DO_WMMA(fa);
        DO_WMMA(fb);
    }

    // D layout: VGPR j, lanes 0-15 -> M=j, lanes 16-31 -> M=j+8; N = lane&15.
    {
        const int tbase = wave * 16 + lh * 8;
#pragma unroll
        for (int j = 0; j < 8; ++j) {
            lds_logits[(tbase + j) * LSTRIDE + 0 * 16 + lm] = acc0[j];
            lds_logits[(tbase + j) * LSTRIDE + 1 * 16 + lm] = acc1[j];
            lds_logits[(tbase + j) * LSTRIDE + 2 * 16 + lm] = acc2[j];
            lds_logits[(tbase + j) * LSTRIDE + 3 * 16 + lm] = acc3[j];
        }
    }
    __syncthreads();

    // ---------------- Softmax + top-2 + per-token loss terms ----------------
    if (tid < TOKENS_PER_BLOCK) {
        const float* lrow = &lds_logits[tid * LSTRIDE];
        float mx = lrow[0];
#pragma unroll 8
        for (int e = 1; e < NUM_EXPERTS; ++e) mx = fmaxf(mx, lrow[e]);

        float s = 0.0f;
        float m0 = -3.0e38f, m1 = -3.0e38f;
        int   i0 = 0, i1 = 0;
#pragma unroll 4
        for (int e = 0; e < NUM_EXPERTS; ++e) {
            float v = lrow[e];
            s += __expf(v - mx);
            if (v > m0)      { m1 = m0; i1 = i0; m0 = v; i0 = e; }
            else if (v > m1) { m1 = v; i1 = e; }
        }
        float logz = mx + __logf(s);
        atomicAdd(&lds_z, logz * logz);

        float e0 = __expf(m0 - mx), e1 = __expf(m1 - mx);
        float rws = 1.0f / (e0 + e1);
        int gt = blockIdx.x * TOKENS_PER_BLOCK + tid;
        out[(size_t)gt * 2 + 0] = e0 * rws;
        out[(size_t)gt * 2 + 1] = e1 * rws;
        float* idxOut = out + (size_t)nTokens * 2;
        idxOut[(size_t)gt * 2 + 0] = (float)i0;
        idxOut[(size_t)gt * 2 + 1] = (float)i1;

        atomicAdd(&lds_cnt[i0], 1.0f);
        atomicAdd(&lds_cnt[i1], 1.0f);

        lds_mx[tid] = mx;
        lds_rs[tid] = 1.0f / s;
    }
    __syncthreads();

    // ---------------- Block -> global reductions (one atomic per expert) ----
    if (tid < NUM_EXPERTS) {
        float ps = 0.0f;
        for (int t = 0; t < TOKENS_PER_BLOCK; ++t)
            ps += __expf(lds_logits[t * LSTRIDE + tid] - lds_mx[t]) * lds_rs[t];
        atomicAdd(&ws[tid], ps);                         // sum of router_probs[:, e]
        atomicAdd(&ws[NUM_EXPERTS + tid], lds_cnt[tid]); // tokens_per_expert
    }
    if (tid == 0) atomicAdd(&ws[NUM_EXPERTS * 2], lds_z);
}

__global__ void finalize_kernel(const float* __restrict__ ws,
                                float* __restrict__ out, int nTokens) {
    if (threadIdx.x == 0 && blockIdx.x == 0) {
        const float invN = 1.0f / (float)nTokens;
        float lb = 0.0f;
        for (int e = 0; e < NUM_EXPERTS; ++e) {
            float P = ws[e] * invN;                      // mean prob
            float f = ws[NUM_EXPERTS + e] * invN * 0.5f; // count / (N*K)
            lb += f * P;
        }
        lb *= (float)NUM_EXPERTS;
        float z = ws[NUM_EXPERTS * 2] * invN;
        out[(size_t)nTokens * 4] = 0.001f * lb + 0.001f * z;
    }
}

extern "C" void kernel_launch(void* const* d_in, const int* in_sizes, int n_in,
                              void* d_out, int out_size, void* d_ws, size_t ws_size,
                              hipStream_t stream) {
    const float* A = (const float*)d_in[0];   // hidden_states [N, 4096]
    const float* W = (const float*)d_in[1];   // W_gate [64, 4096]
    float* out = (float*)d_out;
    float* wsF = (float*)d_ws;
    unsigned int*   WbPairs = (unsigned int*)((char*)d_ws + WB_OFFSET_BYTES);
    unsigned short* Wb      = (unsigned short*)WbPairs;

    const int nTokens = in_sizes[0] / HDIM;   // 16384
    const int nBlocks = nTokens / TOKENS_PER_BLOCK;

    zero_ws_kernel<<<1, 256, 0, stream>>>(wsF);
    convert_w_kernel<<<(NUM_EXPERTS * HDIM / 2 + 255) / 256, 256, 0, stream>>>(W, WbPairs);
    router_gemm_topk_kernel<<<nBlocks, 256, 0, stream>>>(A, Wb, out, wsF, nTokens);
    finalize_kernel<<<1, 64, 0, stream>>>(wsF, out, nTokens);
}
